// CPSFCodebookFused_40707700031514
// MI455X (gfx1250) — matrix operand order repaired and checked by reading, not compiled, not run on hardware
//
#include <hip/hip_runtime.h>
#include <math.h>

typedef __attribute__((ext_vector_type(16))) _Float16 v16h;
typedef __attribute__((ext_vector_type(8)))  float    v8f;

#define WMMAF16(A,B,C) __builtin_amdgcn_wmma_f32_16x16x32_f16(false,(A),false,(B),(short)0,(C),false,false)

#define B_ROWS 4096
#define NDIM   64
#define MCB    8192
#define SDIM   256
#define TINYF  1.17549435e-38f
#define PIF    3.14159265358979323846f

// ---- WMMA fragment address helpers (CDNA5 ISA 7.12.2, wave32) ----
__device__ __forceinline__ int afrag_off(int r, int kp) {
  int lane = r + (((kp >> 3) & 1) << 4);
  int slot = (kp & 7) + ((kp >> 4) << 3);
  return lane * 16 + slot;
}
__device__ __forceinline__ int bfrag_off(int kp, int c) {
  int lane = c + (((kp >> 3) & 1) << 4);
  int slot = (kp & 7) + ((kp >> 4) << 3);
  return lane * 16 + slot;
}

// ---------------- prep: x side (z rows, unit-normalized d rows, |z|^2) ----------------
__global__ void cpsf_prep_x(const float* __restrict__ x_re, const float* __restrict__ x_im,
                            _Float16* __restrict__ xA, float* __restrict__ zsq) {
  int b = blockIdx.x * blockDim.x + threadIdx.x;
  if (b >= B_ROWS) return;
  const float* xr = x_re + (size_t)b * 2 * NDIM;
  const float* xi = x_im + (size_t)b * 2 * NDIM;
  float za = 0.f, da = 0.f;
  for (int k = 0; k < NDIM; ++k) {
    float zr = xr[k], zi = xi[k];
    za += zr * zr + zi * zi;
    float dr = xr[NDIM + k], di = xi[NDIM + k];
    da += dr * dr + di * di;
  }
  zsq[b] = za;
  float n = sqrtf(da);
  if (n == 0.f) n = 1.f;
  float inv = 1.f / n;
  int bt = b >> 4, r = b & 15;
  for (int k = 0; k < NDIM; ++k) {
    int kh = k >> 5, kp = k & 31;
    int off = afrag_off(r, kp);
    xA[(((size_t)bt * 4 + 0) * 2 + kh) * 512 + off] = (_Float16)xr[k];                 // z_re
    xA[(((size_t)bt * 4 + 1) * 2 + kh) * 512 + off] = (_Float16)xi[k];                 // z_im
    xA[(((size_t)bt * 4 + 2) * 2 + kh) * 512 + off] = (_Float16)(xr[NDIM + k] * inv);  // d_re
    xA[(((size_t)bt * 4 + 3) * 2 + kh) * 512 + off] = (_Float16)(xi[NDIM + k] * inv);  // d_im
  }
}

// ---------------- prep: codebook side (5 B-matrices incl. -dj_im; b_par, |z_j|^2) ----------------
__global__ void cpsf_prep_j(const float* __restrict__ zjr, const float* __restrict__ zji,
                            const float* __restrict__ djr, const float* __restrict__ dji,
                            _Float16* __restrict__ jB, float* __restrict__ bpr,
                            float* __restrict__ bpi, float* __restrict__ zjq) {
  int m = blockIdx.x * blockDim.x + threadIdx.x;
  if (m >= MCB) return;
  const float* dr = djr + (size_t)m * NDIM;
  const float* di = dji + (size_t)m * NDIM;
  const float* zr = zjr + (size_t)m * NDIM;
  const float* zi = zji + (size_t)m * NDIM;
  float da = 0.f;
  for (int k = 0; k < NDIM; ++k) da += dr[k] * dr[k] + di[k] * di[k];
  float n = sqrtf(da);
  if (n == 0.f) n = 1.f;
  float inv = 1.f / n;
  float bre = 0.f, bim = 0.f, zq = 0.f;
  int mt = m >> 4, c = m & 15;
  for (int k = 0; k < NDIM; ++k) {
    float dre = dr[k] * inv, dim = di[k] * inv;
    float zre = zr[k], zim = zi[k];
    bre += dre * zre + dim * zim;        // Re(conj(d)*z)
    bim += dre * zim - dim * zre;        // Im(conj(d)*z)
    zq  += zre * zre + zim * zim;
    int kh = k >> 5, kp = k & 31;
    int off = bfrag_off(kp, c);
    jB[(((size_t)mt * 5 + 0) * 2 + kh) * 512 + off] = (_Float16)dre;     // dj_re
    jB[(((size_t)mt * 5 + 1) * 2 + kh) * 512 + off] = (_Float16)dim;     // dj_im
    jB[(((size_t)mt * 5 + 2) * 2 + kh) * 512 + off] = (_Float16)zre;     // zj_re
    jB[(((size_t)mt * 5 + 3) * 2 + kh) * 512 + off] = (_Float16)zim;     // zj_im
    jB[(((size_t)mt * 5 + 4) * 2 + kh) * 512 + off] = (_Float16)(-dim);  // -dj_im
  }
  bpr[m] = bre; bpi[m] = bim; zjq[m] = zq;
}

// ---------------- prep: T_hat -> f16 B fragments, K = m (chunks of 32), N = s (tiles of 16) ----------------
__global__ void cpsf_prep_that(const float* __restrict__ tr, const float* __restrict__ ti,
                               _Float16* __restrict__ tB) {
  int tid = blockIdx.x * blockDim.x + threadIdx.x;
  if (tid >= MCB * SDIM) return;
  int m = tid >> 8, s = tid & 255;
  int mc = m >> 5, kp = m & 31, st = s >> 4, c = s & 15;
  int off = bfrag_off(kp, c);
  tB[(((size_t)mc * 2 + 0) * 16 + st) * 512 + off] = (_Float16)tr[(size_t)m * SDIM + s];
  tB[(((size_t)mc * 2 + 1) * 16 + st) * 512 + off] = (_Float16)ti[(size_t)m * SDIM + s];
}

// ---------------- prep: DFT trig tables (cos, -sin) as B fragments, K = s, N = k ----------------
__global__ void cpsf_prep_trig(_Float16* __restrict__ tg) {
  int tid = blockIdx.x * blockDim.x + threadIdx.x;
  if (tid >= SDIM * SDIM) return;
  int k = tid >> 8, s = tid & 255;
  float ang = (float)((k * s) & 255) * (2.f * PIF / 256.f);  // exact period reduction
  int sc = s >> 5, kp = s & 31, kt = k >> 4, c = k & 15;
  int off = bfrag_off(kp, c);
  tg[((size_t)(0 * 8 + sc) * 16 + kt) * 512 + off] = (_Float16)cosf(ang);
  tg[((size_t)(1 * 8 + sc) * 16 + kt) * 512 + off] = (_Float16)(-sinf(ang));
}

// ---------------- fused main: 4 waves / 16-row B tile; wave w owns chain w and S-quarter w ----------------
__launch_bounds__(128, 1)
__global__ void cpsf_main(const _Float16* __restrict__ xA, const float* __restrict__ zsq,
                          const _Float16* __restrict__ jB, const float* __restrict__ bpr,
                          const float* __restrict__ bpi, const float* __restrict__ zjq,
                          const float* __restrict__ alpha, const float* __restrict__ spar,
                          const float* __restrict__ sperp, const _Float16* __restrict__ tB,
                          const _Float16* __restrict__ trg, float* __restrict__ out) {
  __shared__ float    ldsC[4 * 2 * 256];   // [chain][t][b][n] f32 chain results
  __shared__ _Float16 ldsW[512];           // 16x32 w tile, A-fragment order
  __shared__ _Float16 ldsT[2 * 8 * 512];   // T_re/T_im 16x256 in A-fragment order
  const int tid  = threadIdx.x;
  const int lane = tid & 31;
  const int wid  = tid >> 5;               // wave 0..3 = chain A/B/C/D
  const int hi   = lane >> 4;
  const int ln   = lane & 15;
  const int bt   = blockIdx.x;

  // chain operand selection:
  //  chain0 (c_par re):  z_re.dj_re + z_im.dj_im
  //  chain1 (c_par im):  z_im.dj_re + z_re.(-dj_im)
  //  chain2 (z cross) :  z_re.zj_re + z_im.zj_im
  //  chain3 (d cross) :  d_re.dj_re + d_im.dj_im
  const int a0 = (wid == 1) ? 1 : ((wid == 3) ? 2 : 0);
  const int a1 = (wid == 1) ? 0 : ((wid == 3) ? 3 : 1);
  const int b0 = (wid == 2) ? 2 : 0;
  const int b1 = (wid == 1) ? 4 : ((wid == 2) ? 3 : 1);

  // this wave's two resident A fragments (K=64 -> 2 frags each)
  const v16h* xv = (const v16h*)xA + (size_t)bt * 8 * 32;
  v16h A00 = xv[(a0 * 2 + 0) * 32 + lane], A01 = xv[(a0 * 2 + 1) * 32 + lane];
  v16h A10 = xv[(a1 * 2 + 0) * 32 + lane], A11 = xv[(a1 * 2 + 1) * 32 + lane];

  // epilogue element mapping: 512 elements over 128 threads; n,t fixed per thread
  const int en  = tid & 15;
  const int et  = (tid >> 4) & 1;
  const int ebg = (tid >> 5) & 3;
  float zs4[4];
#pragma unroll
  for (int i = 0; i < 4; ++i) zs4[i] = zsq[bt * 16 + ebg * 4 + i];

  // this wave accumulates S-quarter wid: s-tiles [wid*4, wid*4+4)
  v8f tRe[4] = {};
  v8f tIm[4] = {};

#pragma unroll 1
  for (int ms = 0; ms < MCB / 32; ++ms) {
    // ---- stage 1: this wave's chain for both 16-m tiles ----
#pragma unroll
    for (int t = 0; t < 2; ++t) {
      const int mt = ms * 2 + t;
      const v16h* jv = (const v16h*)jB + (size_t)mt * 10 * 32;
      v16h B00 = jv[(b0 * 2 + 0) * 32 + lane], B01 = jv[(b0 * 2 + 1) * 32 + lane];
      v16h B10 = jv[(b1 * 2 + 0) * 32 + lane], B11 = jv[(b1 * 2 + 1) * 32 + lane];
      v8f acc = {};
      acc = WMMAF16(A00, B00, acc);
      acc = WMMAF16(A01, B01, acc);
      acc = WMMAF16(A10, B10, acc);
      acc = WMMAF16(A11, B11, acc);
#pragma unroll
      for (int v = 0; v < 8; ++v)
        ldsC[((wid * 2 + t) * 16 + v + 8 * hi) * 16 + ln] = acc[v];
    }
    __syncthreads();
    // ---- epilogue: 4 elements per thread, w -> A-fragment in LDS ----
    {
      const int m = (ms * 2 + et) * 16 + en;
      float vbpr = bpr[m], vbpi = bpi[m], vzq = zjq[m];
      float va  = fmaxf(alpha[m], TINYF);
      float vsp = fmaxf(spar[m], TINYF);
      float vse = fmaxf(sperp[m], TINYF);
#pragma unroll
      for (int i = 0; i < 4; ++i) {
        int b = ebg * 4 + i;
        int cbase = et * 256 + b * 16 + en;
        float accA = ldsC[0 * 512 + cbase];
        float accB = ldsC[1 * 512 + cbase];
        float accC = ldsC[2 * 512 + cbase];
        float accD = ldsC[3 * 512 + cbase];
        float cre = accA - vbpr;
        float cim = accB - vbpi;
        float parsq = cre * cre + cim * cim;
        float tot = zs4[i] + vzq - 2.f * accC;
        float perp = fmaxf(tot - parsq, 0.f);
        float dd = fmaxf(2.f - 2.f * accD, 0.f);  // both d vectors unit norm
        float q = parsq / vsp + (perp + dd) / vse;
        float w = va * __expf(-PIF * q);
        ldsW[afrag_off(b, en + 16 * et)] = (_Float16)w;
      }
    }
    __syncthreads();
    // ---- stage 2: this wave's S-quarter against full 16x32 w tile ----
    v16h wa = ((const v16h*)ldsW)[lane];
    const v16h* tv = (const v16h*)tB + (size_t)ms * 2 * 16 * 32;
    __builtin_prefetch(tv + 2 * 16 * 32, 0, 0);  // next K-chunk of T_hat fragments
#pragma unroll
    for (int st = 0; st < 4; ++st) {
      const int stg = wid * 4 + st;
      v16h br = tv[(0 * 16 + stg) * 32 + lane];
      tRe[st] = WMMAF16(wa, br, tRe[st]);
      v16h bi = tv[(1 * 16 + stg) * 32 + lane];
      tIm[st] = WMMAF16(wa, bi, tIm[st]);
    }
  }
  __syncthreads();

  // T accumulators (C-layout) -> LDS A-fragments over K=S; each wave dumps its S-quarter
#pragma unroll
  for (int st = 0; st < 4; ++st) {
    const int s = (wid * 4 + st) * 16 + ln;
    const int sc = s >> 5, kp = s & 31;
#pragma unroll
    for (int v = 0; v < 8; ++v) {
      int off = afrag_off(v + 8 * hi, kp);
      ldsT[(0 * 8 + sc) * 512 + off] = (_Float16)tRe[st][v];
      ldsT[(1 * 8 + sc) * 512 + off] = (_Float16)tIm[st][v];
    }
  }
  __syncthreads();

  // fused DFT: out[b,k] = sum_s T_re*cos + T_im*(-sin); wave wid produces k-tiles [wid*4, wid*4+4)
#pragma unroll 1
  for (int kt = 0; kt < 4; ++kt) {
    const int ktg = wid * 4 + kt;
    v8f o = {};
#pragma unroll
    for (int sc = 0; sc < 8; ++sc) {
      v16h ta = ((const v16h*)ldsT)[(0 * 8 + sc) * 32 + lane];
      v16h cb = ((const v16h*)trg)[((size_t)(0 * 8 + sc) * 16 + ktg) * 32 + lane];
      o = WMMAF16(ta, cb, o);
      v16h tb = ((const v16h*)ldsT)[(1 * 8 + sc) * 32 + lane];
      v16h sb = ((const v16h*)trg)[((size_t)(1 * 8 + sc) * 16 + ktg) * 32 + lane];
      o = WMMAF16(tb, sb, o);
    }
#pragma unroll
    for (int v = 0; v < 8; ++v) {
      out[(size_t)(bt * 16 + v + 8 * hi) * SDIM + ktg * 16 + ln] = o[v];
    }
  }
}

extern "C" void kernel_launch(void* const* d_in, const int* in_sizes, int n_in,
                              void* d_out, int out_size, void* d_ws, size_t ws_size,
                              hipStream_t stream) {
  (void)in_sizes; (void)n_in; (void)out_size; (void)ws_size;
  const float* x_re  = (const float*)d_in[0];
  const float* x_im  = (const float*)d_in[1];
  const float* zj_re = (const float*)d_in[2];
  const float* zj_im = (const float*)d_in[3];
  const float* dj_re = (const float*)d_in[4];
  const float* dj_im = (const float*)d_in[5];
  const float* th_re = (const float*)d_in[6];
  const float* th_im = (const float*)d_in[7];
  const float* alpha = (const float*)d_in[8];
  const float* spar  = (const float*)d_in[9];
  const float* sperp = (const float*)d_in[10];
  float* out = (float*)d_out;

  char* ws = (char*)d_ws;
  _Float16* xA  = (_Float16*)(ws + 0);          //  2 MiB: 256 tiles * 4 mats * 2 kh * 512 halves
  float*    zsq = (float*)(ws + 2097152);       // 16 KiB
  _Float16* jB  = (_Float16*)(ws + 2113536);    //  5 MiB: 512 tiles * 5 mats * 2 kh * 512 halves
  float*    bpr = (float*)(ws + 7356416);       // 32 KiB
  float*    bpi = (float*)(ws + 7389184);       // 32 KiB
  float*    zjq = (float*)(ws + 7421952);       // 32 KiB
  _Float16* tB  = (_Float16*)(ws + 7454720);    //  8 MiB: 256 mchunks * 2 mats * 16 stiles * 512
  _Float16* trg = (_Float16*)(ws + 15843328);   // 256 KiB: 2 mats * 8 schunks * 16 ktiles * 512

  cpsf_prep_x<<<B_ROWS / 256, 256, 0, stream>>>(x_re, x_im, xA, zsq);
  cpsf_prep_j<<<MCB / 256, 256, 0, stream>>>(zj_re, zj_im, dj_re, dj_im, jB, bpr, bpi, zjq);
  cpsf_prep_that<<<(MCB * SDIM) / 256, 256, 0, stream>>>(th_re, th_im, tB);
  cpsf_prep_trig<<<(SDIM * SDIM) / 256, 256, 0, stream>>>(trg);
  cpsf_main<<<B_ROWS / 16, 128, 0, stream>>>(xA, zsq, jB, bpr, bpi, zjq, alpha, spar, sperp,
                                             tB, trg, out);
}
